// TopologyMessageLayer_90752658964793
// MI455X (gfx1250) — matrix operand order repaired and checked by reading, not compiled, not run on hardware
//
#include <hip/hip_runtime.h>

#define cB  8
#define cNF 16384
#define cNE 32768
#define cD  128

typedef __attribute__((ext_vector_type(16))) __bf16 v16bf;
typedef __attribute__((ext_vector_type(8)))  __bf16 v8bf;
typedef __attribute__((ext_vector_type(4)))  __bf16 v4bf;
typedef __attribute__((ext_vector_type(8)))  float  v8f;

__device__ __forceinline__ v8f wmma_bf16(v16bf a, v16bf b, v8f c) {
  return __builtin_amdgcn_wmma_f32_16x16x32_bf16(false, a, false, b, (short)0, c, false, false);
}

// 16x32 A-fragment straight from bf16 LDS (two ds_load_b128, no conversion).
// Lane l<16: row l, K = kt*32 + {0..7, 16..23}; lane l>=16: row l-16, K = kt*32 + {8..15, 24..31}.
__device__ __forceinline__ v16bf a_frag_bf(const __bf16* base, int stride, int mt, int kt) {
  int lane = threadIdx.x & 31;
  int row  = (mt << 4) + (lane & 15);
  int kb   = (kt << 5) + ((lane >> 4) << 3);
  const __bf16* p = base + row * stride + kb;
  v8bf lo = *(const v8bf*)p;
  v8bf hi = *(const v8bf*)(p + 16);
  v16bf a;
  #pragma unroll
  for (int i = 0; i < 8; ++i) { a[i] = lo[i]; a[8 + i] = hi[i]; }
  return a;
}

// 16x32 A-fragment from fp32 LDS rows with on-the-fly bf16 conversion.
__device__ __forceinline__ v16bf a_frag_f32(const float* base, int stride, int mt, int kt) {
  int lane = threadIdx.x & 31;
  int row  = (mt << 4) + (lane & 15);
  int kb   = (kt << 5) + ((lane >> 4) << 3);
  const float* p = base + row * stride + kb;
  const float4* p0 = (const float4*)p;
  const float4* p1 = (const float4*)(p + 16);
  float4 x0 = p0[0], x1 = p0[1], y0 = p1[0], y1 = p1[1];
  v16bf a;
  a[0]=(__bf16)x0.x; a[1]=(__bf16)x0.y; a[2]=(__bf16)x0.z; a[3]=(__bf16)x0.w;
  a[4]=(__bf16)x1.x; a[5]=(__bf16)x1.y; a[6]=(__bf16)x1.z; a[7]=(__bf16)x1.w;
  a[8]=(__bf16)y0.x; a[9]=(__bf16)y0.y; a[10]=(__bf16)y0.z; a[11]=(__bf16)y0.w;
  a[12]=(__bf16)y1.x; a[13]=(__bf16)y1.y; a[14]=(__bf16)y1.z; a[15]=(__bf16)y1.w;
  return a;
}

// Pre-packed B fragments: 1024B per (ktile,ntile) fragment, 32B contiguous per lane.
__device__ __forceinline__ v16bf b_frag(const __bf16* packed, int frag) {
  return *(const v16bf*)(packed + ((size_t)frag << 9) + ((size_t)(threadIdx.x & 31) << 4));
}

__device__ __forceinline__ float gelu_exact(float x) { return 0.5f * x * (1.f + erff(x * 0.70710678118f)); }
__device__ __forceinline__ float sigmoidf(float x)   { return 1.f / (1.f + expf(-x)); }

__device__ __forceinline__ v4bf cvt4(float4 x) {
  v4bf r; r[0]=(__bf16)x.x; r[1]=(__bf16)x.y; r[2]=(__bf16)x.z; r[3]=(__bf16)x.w; return r;
}

// Repack fp32 (K,N) weight into bf16 WMMA-B fragments: out[(kt*ntiles+nt)*512 + lane*16 + h]
__global__ __launch_bounds__(256) void pack_b(const float* __restrict__ W, __bf16* __restrict__ out,
                                              int K, int N) {
  int idx = blockIdx.x * 256 + threadIdx.x;
  if (idx >= K * N) return;
  int f = idx >> 9, e = idx & 511;
  int lane = e >> 4, h = e & 15;
  int ntiles = N >> 4;
  int kt = f / ntiles, nt = f - kt * ntiles;
  int n = (nt << 4) + (lane & 15);
  int k = (kt << 5) + ((lane >= 16) ? 8 : 0) + ((h < 8) ? h : h + 8);
  out[idx] = (__bf16)W[k * N + n];
}

__global__ __launch_bounds__(256) void zero_kernel(float* __restrict__ outF, float* __restrict__ counts) {
  size_t n1 = (size_t)cB * cNF * cD;
  size_t total = n1 + (size_t)cB * cNF;
  for (size_t i = (size_t)blockIdx.x * 256 + threadIdx.x; i < total; i += (size_t)gridDim.x * 256) {
    if (i < n1) outF[i] = 0.f; else counts[i - n1] = 0.f;
  }
}

// ---------------- Edge update: 32 edges per workgroup, 8 waves, B reused x2 ----------------
__global__ __launch_bounds__(256) void edge_kernel(
    const float* __restrict__ Ff, const float* __restrict__ Ef,
    const int* __restrict__ etf, const unsigned char* __restrict__ emask,
    const __bf16* __restrict__ Wfe1, const __bf16* __restrict__ Wfe2, const __bf16* __restrict__ Wge,
    const float* __restrict__ bfe1, const float* __restrict__ bfe2, const float* __restrict__ bge,
    const float* __restrict__ ge, const float* __restrict__ be,
    float* __restrict__ outE) {
  __shared__ float  Ef32[32 * 128];    // E rows fp32 (WMMA A via cvt + residual base; resid in-place)
  __shared__ __bf16 f12Bf[32 * 264];   // [F[f1] | F[f2]] bf16, cols 0..255
  __shared__ __bf16 h1Bf[32 * 264];    // layer1 activations bf16, cols 0..255
  __shared__ __bf16 msgBf[32 * 136];   // masked edge message bf16, cols 0..127
  __shared__ int    f1s[32], f2s[32];
  __shared__ float  vfs[32];

  int bb = blockIdx.x >> 10;            // cNE/32 = 1024 tiles per batch
  int e0 = (blockIdx.x & 1023) << 5;
  int lane = threadIdx.x & 31;
  int w = threadIdx.x >> 5;
  int col = lane & 15;
  int rb = (lane >> 4) << 3;

  if (threadIdx.x < 32) {
    size_t e = (size_t)bb * cNE + e0 + threadIdx.x;
    int i0 = etf[e * 2 + 0], i1 = etf[e * 2 + 1];
    f1s[threadIdx.x] = min(max(i0, 0), cNF - 1);
    f2s[threadIdx.x] = min(max(i1, 0), cNF - 1);
    vfs[threadIdx.x] = (emask[e] && i0 >= 0 && i1 >= 0) ? 1.f : 0.f;
  }
  __syncthreads();

  // Gather: E rows -> fp32 LDS; F[f1],F[f2] rows -> bf16 LDS
  for (int i = threadIdx.x; i < 3072; i += 256) {
    int r = i / 96;
    int c = (i - r * 96) << 2;
    if (c < 128) {
      float4 x = *(const float4*)(Ef + ((size_t)bb * cNE + e0 + r) * cD + c);
      *(float4*)&Ef32[r * 128 + c] = x;
    } else {
      const float* src = (c < 256) ? (Ff + ((size_t)bb * cNF + f1s[r]) * cD + (c - 128))
                                   : (Ff + ((size_t)bb * cNF + f2s[r]) * cD + (c - 256));
      *(v4bf*)&f12Bf[r * 264 + (c - 128)] = cvt4(*(const float4*)src);
    }
  }
  __syncthreads();

  // Layer1: (32x384)@(384x256) + bias + exact GELU. kt-outer: B fragment reused across 2 M-tiles.
  {
    v8f acc[2][2] = {};
    #pragma unroll
    for (int kt = 0; kt < 12; ++kt) {
      v16bf b0 = b_frag(Wfe1, kt * 16 + (w << 1));
      v16bf b1 = b_frag(Wfe1, kt * 16 + (w << 1) + 1);
      #pragma unroll
      for (int mt = 0; mt < 2; ++mt) {
        v16bf a = (kt < 4) ? a_frag_f32(Ef32, 128, mt, kt) : a_frag_bf(f12Bf, 264, mt, kt - 4);
        acc[mt][0] = wmma_bf16(a, b0, acc[mt][0]);
        acc[mt][1] = wmma_bf16(a, b1, acc[mt][1]);
      }
    }
    #pragma unroll
    for (int t2 = 0; t2 < 2; ++t2) {
      int gc = (((w << 1) + t2) << 4) + col;
      float bias = bfe1[gc];
      #pragma unroll
      for (int mt = 0; mt < 2; ++mt)
        #pragma unroll
        for (int v = 0; v < 8; ++v)
          h1Bf[((mt << 4) + rb + v) * 264 + gc] = (__bf16)gelu_exact(acc[mt][t2][v] + bias);
    }
  }
  __syncthreads();

  // Layer2: (32x256)@(256x128) + bias, masked by validity
  float mreg[2][8];
  {
    v8f acc[2] = {};
    #pragma unroll
    for (int kt = 0; kt < 8; ++kt) {
      v16bf b = b_frag(Wfe2, kt * 8 + w);
      #pragma unroll
      for (int mt = 0; mt < 2; ++mt)
        acc[mt] = wmma_bf16(a_frag_bf(h1Bf, 264, mt, kt), b, acc[mt]);
    }
    int gc = (w << 4) + col;
    float bias = bfe2[gc];
    #pragma unroll
    for (int mt = 0; mt < 2; ++mt)
      #pragma unroll
      for (int v = 0; v < 8; ++v) {
        int r = (mt << 4) + rb + v;
        float m = (acc[mt][v] + bias) * vfs[r];
        mreg[mt][v] = m;
        msgBf[r * 136 + gc] = (__bf16)m;
      }
  }
  __syncthreads();

  // Gate on [E | msg] + residual (in-place into Ef32 after all WMMA reads complete)
  {
    v8f acc[2] = {};
    #pragma unroll
    for (int kt = 0; kt < 8; ++kt) {
      v16bf b = b_frag(Wge, kt * 8 + w);
      #pragma unroll
      for (int mt = 0; mt < 2; ++mt) {
        v16bf a = (kt < 4) ? a_frag_f32(Ef32, 128, mt, kt) : a_frag_bf(msgBf, 136, mt, kt - 4);
        acc[mt] = wmma_bf16(a, b, acc[mt]);
      }
    }
    __syncthreads();   // all waves done reading Ef32 before in-place residual writes
    int gc = (w << 4) + col;
    float bias = bge[gc];
    #pragma unroll
    for (int mt = 0; mt < 2; ++mt)
      #pragma unroll
      for (int v = 0; v < 8; ++v) {
        int r = (mt << 4) + rb + v;
        float g = sigmoidf(acc[mt][v] + bias);
        Ef32[r * 128 + gc] += g * mreg[mt][v];
      }
  }
  __syncthreads();

  // LayerNorm: wave w handles rows 4w..4w+3; 4 elems/lane + wave32 shuffle reduce
  #pragma unroll
  for (int rr = 0; rr < 4; ++rr) {
    int r = (w << 2) + rr;
    float4 x = *(const float4*)&Ef32[r * 128 + lane * 4];
    float s  = x.x + x.y + x.z + x.w;
    float sq = x.x * x.x + x.y * x.y + x.z * x.z + x.w * x.w;
    #pragma unroll
    for (int m = 16; m > 0; m >>= 1) { s += __shfl_xor(s, m, 32); sq += __shfl_xor(sq, m, 32); }
    float mean = s * (1.f / 128.f);
    float var  = sq * (1.f / 128.f) - mean * mean;
    float inv  = rsqrtf(var + 1e-5f);
    float4 gv = *(const float4*)(ge + lane * 4);
    float4 bv = *(const float4*)(be + lane * 4);
    float4 o;
    o.x = (x.x - mean) * inv * gv.x + bv.x;
    o.y = (x.y - mean) * inv * gv.y + bv.y;
    o.z = (x.z - mean) * inv * gv.z + bv.z;
    o.w = (x.w - mean) * inv * gv.w + bv.w;
    *(float4*)(outE + ((size_t)bb * cNE + e0 + r) * cD + lane * 4) = o;
  }
}

// ---------------- Scatter E_new -> face accumulators (fp32 atomics) ----------------
__global__ __launch_bounds__(256) void scatter_kernel(
    const float* __restrict__ outE, const int* __restrict__ etf,
    const unsigned char* __restrict__ emask,
    float* __restrict__ faceMsg, float* __restrict__ counts) {
  size_t gid = (size_t)blockIdx.x * 256 + threadIdx.x;  // B*NE*32 threads
  int c4 = (int)(gid & 31);
  size_t e = gid >> 5;
  int bb = (int)(e >> 15);  // cNE = 2^15
  int i0 = etf[e * 2 + 0], i1 = etf[e * 2 + 1];
  if (!(emask[e] && i0 >= 0 && i1 >= 0)) return;
  int f1c = min(max(i0, 0), cNF - 1);
  int f2c = min(max(i1, 0), cNF - 1);
  float4 v = *(const float4*)(outE + e * cD + c4 * 4);
  float* p1 = faceMsg + ((size_t)bb * cNF + f1c) * cD + c4 * 4;
  float* p2 = faceMsg + ((size_t)bb * cNF + f2c) * cD + c4 * 4;
  atomicAdd(p1 + 0, v.x); atomicAdd(p1 + 1, v.y); atomicAdd(p1 + 2, v.z); atomicAdd(p1 + 3, v.w);
  atomicAdd(p2 + 0, v.x); atomicAdd(p2 + 1, v.y); atomicAdd(p2 + 2, v.z); atomicAdd(p2 + 3, v.w);
  if (c4 == 0) {
    atomicAdd(counts + (size_t)bb * cNF + f1c, 1.f);
    atomicAdd(counts + (size_t)bb * cNF + f2c, 1.f);
  }
}

// ---------------- Face update: 32 faces per workgroup, 8 waves, B reused x2 ----------------
__global__ __launch_bounds__(256) void face_kernel(
    const float* __restrict__ Ff, float* __restrict__ outF,
    const float* __restrict__ counts, const unsigned char* __restrict__ fmask,
    const __bf16* __restrict__ Wef1, const __bf16* __restrict__ Wef2, const __bf16* __restrict__ Wgf,
    const float* __restrict__ bef1, const float* __restrict__ bef2, const float* __restrict__ bgf,
    const float* __restrict__ gf, const float* __restrict__ bf) {
  __shared__ float  Ff32[32 * 128];    // F rows fp32 (A via cvt + residual base; resid in-place)
  __shared__ __bf16 fmBf[32 * 136];    // face_msg mean bf16
  __shared__ __bf16 hBf[32 * 136];     // ef1 activations bf16
  __shared__ __bf16 updBf[32 * 136];   // ef2 output bf16 (unmasked)
  __shared__ float  cntS[32];
  __shared__ float  fms[32];

  int bb = blockIdx.x >> 9;            // cNF/32 = 512 tiles per batch
  int f0 = (blockIdx.x & 511) << 5;
  int lane = threadIdx.x & 31;
  int w = threadIdx.x >> 5;
  int col = lane & 15;
  int rb = (lane >> 4) << 3;

  if (threadIdx.x < 32) {
    size_t f = (size_t)bb * cNF + f0 + threadIdx.x;
    cntS[threadIdx.x] = counts[f] + 1e-8f;
    fms[threadIdx.x]  = fmask[f] ? 1.f : 0.f;
  }
  __syncthreads();

  // Load F rows (fp32) and face_msg/cnt (bf16). face_msg lives in outF region (read-before-write).
  for (int i = threadIdx.x; i < 2048; i += 256) {
    int r = i >> 6;
    int c = (i & 63) << 2;
    if (c < 128) {
      *(float4*)&Ff32[r * 128 + c] = *(const float4*)(Ff + ((size_t)bb * cNF + f0 + r) * cD + c);
    } else {
      float4 x = *(const float4*)(outF + ((size_t)bb * cNF + f0 + r) * cD + (c - 128));
      float inv = 1.f / cntS[r];
      x.x *= inv; x.y *= inv; x.z *= inv; x.w *= inv;
      *(v4bf*)&fmBf[r * 136 + (c - 128)] = cvt4(x);
    }
  }
  __syncthreads();

  // ef1: (32x256)@(256x128) + bias + GELU
  {
    v8f acc[2] = {};
    #pragma unroll
    for (int kt = 0; kt < 8; ++kt) {
      v16bf b = b_frag(Wef1, kt * 8 + w);
      #pragma unroll
      for (int mt = 0; mt < 2; ++mt) {
        v16bf a = (kt < 4) ? a_frag_f32(Ff32, 128, mt, kt) : a_frag_bf(fmBf, 136, mt, kt - 4);
        acc[mt] = wmma_bf16(a, b, acc[mt]);
      }
    }
    int gc = (w << 4) + col;
    float bias = bef1[gc];
    #pragma unroll
    for (int mt = 0; mt < 2; ++mt)
      #pragma unroll
      for (int v = 0; v < 8; ++v)
        hBf[((mt << 4) + rb + v) * 136 + gc] = (__bf16)gelu_exact(acc[mt][v] + bias);
  }
  __syncthreads();

  // ef2: (32x128)@(128x128) + bias (unmasked update)
  float ureg[2][8];
  {
    v8f acc[2] = {};
    #pragma unroll
    for (int kt = 0; kt < 4; ++kt) {
      v16bf b = b_frag(Wef2, kt * 8 + w);
      #pragma unroll
      for (int mt = 0; mt < 2; ++mt)
        acc[mt] = wmma_bf16(a_frag_bf(hBf, 136, mt, kt), b, acc[mt]);
    }
    int gc = (w << 4) + col;
    float bias = bef2[gc];
    #pragma unroll
    for (int mt = 0; mt < 2; ++mt)
      #pragma unroll
      for (int v = 0; v < 8; ++v) {
        float u = acc[mt][v] + bias;
        ureg[mt][v] = u;
        updBf[((mt << 4) + rb + v) * 136 + gc] = (__bf16)u;
      }
  }
  __syncthreads();

  // Gate on [F | upd] + masked residual (in-place into Ff32)
  {
    v8f acc[2] = {};
    #pragma unroll
    for (int kt = 0; kt < 8; ++kt) {
      v16bf b = b_frag(Wgf, kt * 8 + w);
      #pragma unroll
      for (int mt = 0; mt < 2; ++mt) {
        v16bf a = (kt < 4) ? a_frag_f32(Ff32, 128, mt, kt) : a_frag_bf(updBf, 136, mt, kt - 4);
        acc[mt] = wmma_bf16(a, b, acc[mt]);
      }
    }
    __syncthreads();   // all waves done reading Ff32 before in-place residual writes
    int gc = (w << 4) + col;
    float bias = bgf[gc];
    #pragma unroll
    for (int mt = 0; mt < 2; ++mt)
      #pragma unroll
      for (int v = 0; v < 8; ++v) {
        int r = (mt << 4) + rb + v;
        float g = sigmoidf(acc[mt][v] + bias);
        Ff32[r * 128 + gc] += g * ureg[mt][v] * fms[r];
      }
  }
  __syncthreads();

  // LayerNorm -> F_new (overwrites face_msg region of d_out)
  #pragma unroll
  for (int rr = 0; rr < 4; ++rr) {
    int r = (w << 2) + rr;
    float4 x = *(const float4*)&Ff32[r * 128 + lane * 4];
    float s  = x.x + x.y + x.z + x.w;
    float sq = x.x * x.x + x.y * x.y + x.z * x.z + x.w * x.w;
    #pragma unroll
    for (int m = 16; m > 0; m >>= 1) { s += __shfl_xor(s, m, 32); sq += __shfl_xor(sq, m, 32); }
    float mean = s * (1.f / 128.f);
    float var  = sq * (1.f / 128.f) - mean * mean;
    float inv  = rsqrtf(var + 1e-5f);
    float4 gv = *(const float4*)(gf + lane * 4);
    float4 bv = *(const float4*)(bf + lane * 4);
    float4 o;
    o.x = (x.x - mean) * inv * gv.x + bv.x;
    o.y = (x.y - mean) * inv * gv.y + bv.y;
    o.z = (x.z - mean) * inv * gv.z + bv.z;
    o.w = (x.w - mean) * inv * gv.w + bv.w;
    *(float4*)(outF + ((size_t)bb * cNF + f0 + r) * cD + lane * 4) = o;
  }
}

extern "C" void kernel_launch(void* const* d_in, const int* in_sizes, int n_in,
                              void* d_out, int out_size, void* d_ws, size_t ws_size,
                              hipStream_t stream) {
  (void)in_sizes; (void)n_in; (void)out_size; (void)ws_size;
  const float* Ff   = (const float*)d_in[0];
  const float* Ef   = (const float*)d_in[1];
  const int*   etf  = (const int*)d_in[2];
  const unsigned char* fmask = (const unsigned char*)d_in[3];
  const unsigned char* emask = (const unsigned char*)d_in[4];
  const float* w_fe1 = (const float*)d_in[5];  const float* b_fe1 = (const float*)d_in[6];
  const float* w_fe2 = (const float*)d_in[7];  const float* b_fe2 = (const float*)d_in[8];
  const float* w_ge  = (const float*)d_in[9];  const float* b_ge  = (const float*)d_in[10];
  const float* g_e   = (const float*)d_in[11]; const float* be_e  = (const float*)d_in[12];
  const float* w_ef1 = (const float*)d_in[13]; const float* b_ef1 = (const float*)d_in[14];
  const float* w_ef2 = (const float*)d_in[15]; const float* b_ef2 = (const float*)d_in[16];
  const float* w_gf  = (const float*)d_in[17]; const float* b_gf  = (const float*)d_in[18];
  const float* g_f   = (const float*)d_in[19]; const float* be_f  = (const float*)d_in[20];

  // Workspace layout: packed bf16 weights (491520 B), then fp32 counts at +512KB.
  __bf16* wsb  = (__bf16*)d_ws;
  __bf16* Wfe1 = wsb;            // 98304 elems
  __bf16* Wfe2 = wsb + 98304;    // 32768
  __bf16* Wge  = wsb + 131072;   // 32768
  __bf16* Wef1 = wsb + 163840;   // 32768
  __bf16* Wef2 = wsb + 196608;   // 16384
  __bf16* Wgf  = wsb + 212992;   // 32768
  float* counts = (float*)((char*)d_ws + 524288);  // cB*cNF floats

  float* outF = (float*)d_out;
  float* outE = outF + (size_t)cB * cNF * cD;

  pack_b<<<(98304 + 255) / 256, 256, 0, stream>>>(w_fe1, Wfe1, 384, 256);
  pack_b<<<(32768 + 255) / 256, 256, 0, stream>>>(w_fe2, Wfe2, 256, 128);
  pack_b<<<(32768 + 255) / 256, 256, 0, stream>>>(w_ge,  Wge,  256, 128);
  pack_b<<<(32768 + 255) / 256, 256, 0, stream>>>(w_ef1, Wef1, 256, 128);
  pack_b<<<(16384 + 255) / 256, 256, 0, stream>>>(w_ef2, Wef2, 128, 128);
  pack_b<<<(32768 + 255) / 256, 256, 0, stream>>>(w_gf,  Wgf,  256, 128);

  zero_kernel<<<4096, 256, 0, stream>>>(outF, counts);

  edge_kernel<<<cB * (cNE / 32), 256, 0, stream>>>(
      Ff, Ef, etf, emask, Wfe1, Wfe2, Wge, b_fe1, b_fe2, b_ge, g_e, be_e, outE);

  scatter_kernel<<<(cB * cNE * 32) / 256, 256, 0, stream>>>(outE, etf, emask, outF, counts);

  face_kernel<<<cB * (cNF / 32), 256, 0, stream>>>(
      Ff, outF, counts, fmask, Wef1, Wef2, Wgf, b_ef1, b_ef2, b_gf, g_f, be_f);
}
